// Net_13211319403055
// MI455X (gfx1250) — matrix operand order repaired and checked by reading, not compiled, not run on hardware
//
#include <hip/hip_runtime.h>
#include <hip/hip_bf16.h>
#include <math.h>

// ---------------- CDNA5 WMMA helpers (wave32, 16x16x32 f16 -> f32) ----------
typedef __attribute__((ext_vector_type(16))) _Float16 v16h;
typedef __attribute__((ext_vector_type(8)))  _Float16 v8h;
typedef __attribute__((ext_vector_type(8)))  float    v8f;

__device__ __forceinline__ v8f wmma16(v16h a, v16h b, v8f c) {
  // (neg_a, A, neg_b, B, c_mod, C, reuse_a, reuse_b)
  return __builtin_amdgcn_wmma_f32_16x16x32_f16(false, a, false, b, (short)0, c,
                                                false, false);
}

// A fragment: 16x32 f16 tile, row-major source, row stride `ld` (elements).
// ISA layout (7.12.2): lane m=lane&15, half kh=lane>>4;
//   regs 0..7  <- K = kh*8 + 0..7      regs 8..15 <- K = 16 + kh*8 + 0..7
__device__ __forceinline__ v16h load_fragA(const _Float16* base, int ld, int lane) {
  const _Float16* p = base + (lane & 15) * ld + ((lane >> 4) & 1) * 8;
  v16h r;
  ((v8h*)&r)[0] = *(const v8h*)(p);
  ((v8h*)&r)[1] = *(const v8h*)(p + 16);
  return r;
}

// B fragment: 32x16 f16. Source is B^T stored row-major [N][K] (row stride ld).
// Same K order as A: regs 0..7 <- K = kh*8 + 0..7, regs 8..15 <- K = 16 + kh*8 + 0..7.
__device__ __forceinline__ v16h load_fragB(const _Float16* bt, int ld, int lane) {
  const _Float16* p = bt + (lane & 15) * ld + ((lane >> 4) & 1) * 8;
  v16h r;
  ((v8h*)&r)[0] = *(const v8h*)(p);
  ((v8h*)&r)[1] = *(const v8h*)(p + 16);
  return r;
}

// Branch-free ELU using the hardware transcendental (v_exp_f32) instead of
// the branchy libm expm1f polynomial.
__device__ __forceinline__ float elu1(float x) {
  return x > 0.0f ? x : (__expf(x) - 1.0f);
}

// Sum of squares of a row of `n8*8` f16 elements (vectorized LDS reads).
__device__ __forceinline__ float row_sumsq(const _Float16* row, int n8) {
  float s = 0.0f;
  for (int c = 0; c < n8; ++c) {
    v8h v = *(const v8h*)(row + c * 8);
#pragma unroll
    for (int e = 0; e < 8; ++e) { float f = (float)v[e]; s += f * f; }
  }
  return s;
}

// ---------------- weight prep -----------------------------------------------
// Wt[n][k] = (k < K) ? W[k][n] : 0   (f32 -> f16, transposed, K padded)
__global__ void transpose_pad_f16(const float* __restrict__ W, int K, int N,
                                  int Kpad, _Float16* __restrict__ Wt) {
  int total = N * Kpad;
  for (int i = blockIdx.x * blockDim.x + threadIdx.x; i < total;
       i += gridDim.x * blockDim.x) {
    int n = i / Kpad, k = i % Kpad;
    float v = (k < K) ? W[k * N + n] : 0.0f;
    Wt[n * Kpad + k] = (_Float16)v;
  }
}

// EdgeConv weight split: [xi, xj-xi]@W == xi@(Wtop-Wbot) + xj@Wbot
// W is (256,128); outputs are transposed (128x128) f16.
__global__ void combine_edge_f16(const float* __restrict__ W,
                                 _Float16* __restrict__ WaT,
                                 _Float16* __restrict__ WbT) {
  int i = blockIdx.x * blockDim.x + threadIdx.x;
  if (i < 128 * 128) {
    int n = i / 128, k = i % 128;
    float top = W[k * 128 + n];
    float bot = W[(128 + k) * 128 + n];
    WaT[n * 128 + k] = (_Float16)(top - bot);
    WbT[n * 128 + k] = (_Float16)bot;
  }
}

// ---------------- node MLPs: relu(elu(X@W1+b1)@W2+b2) -----------------------
// 128 rows per block, 8 waves, each wave owns a 16-row strip. N = 128.
__global__ __launch_bounds__(256) void mlp2_kernel(
    const float* __restrict__ X, int inD, const _Float16* __restrict__ W1t,
    const float* __restrict__ b1, const _Float16* __restrict__ W2t,
    const float* __restrict__ b2, _Float16* __restrict__ outF) {
  __shared__ _Float16 Xp[128 * 40];   // K padded to 32, stride 40 (bank stagger)
  __shared__ _Float16 H[128 * 136];   // hidden, stride 136 (conflict-free cols)
  int tid = threadIdx.x, lane = tid & 31, wave = tid >> 5;
  long row0 = (long)blockIdx.x * 128;

  for (int i = tid; i < 128 * 32; i += 256) {
    int r = i >> 5, c = i & 31;
    Xp[r * 40 + c] = (c < inD) ? (_Float16)X[(row0 + r) * inD + c] : (_Float16)0.0f;
  }
  __syncthreads();

  int n_lo = lane & 15, mh = (lane >> 4) << 3;
  // layer 1: K=32 (padded), one WMMA per n-tile
  {
    v16h a = load_fragA(&Xp[(wave * 16) * 40], 40, lane);
    for (int nt = 0; nt < 8; ++nt) {
      v16h b = load_fragB(W1t + nt * 16 * 32, 32, lane);
      v8f c = {};
      c = wmma16(a, b, c);
      int n = nt * 16 + n_lo;
      float bias = b1[n];
#pragma unroll
      for (int j = 0; j < 8; ++j) {
        int m = wave * 16 + j + mh;
        H[m * 136 + n] = (_Float16)elu1(c[j] + bias);
      }
    }
  }
  __syncthreads();
  // layer 2: K=128, chain 4 WMMAs
  {
    v16h a0 = load_fragA(&H[(wave * 16) * 136 + 0], 136, lane);
    v16h a1 = load_fragA(&H[(wave * 16) * 136 + 32], 136, lane);
    v16h a2 = load_fragA(&H[(wave * 16) * 136 + 64], 136, lane);
    v16h a3 = load_fragA(&H[(wave * 16) * 136 + 96], 136, lane);
    for (int nt = 0; nt < 8; ++nt) {
      v8f c = {};
      c = wmma16(a0, load_fragB(W2t + nt * 16 * 128 + 0, 128, lane), c);
      c = wmma16(a1, load_fragB(W2t + nt * 16 * 128 + 32, 128, lane), c);
      c = wmma16(a2, load_fragB(W2t + nt * 16 * 128 + 64, 128, lane), c);
      c = wmma16(a3, load_fragB(W2t + nt * 16 * 128 + 96, 128, lane), c);
      int n = nt * 16 + n_lo;
      float bias = b2[n];
#pragma unroll
      for (int j = 0; j < 8; ++j) {
        int m = wave * 16 + j + mh;
        float v = c[j] + bias;
        outF[(row0 + m) * 128 + n] = (_Float16)(v > 0.0f ? v : 0.0f);
      }
    }
  }
}

// ---------------- EdgeConv 1: src=sv(16), dst=trk(64), K=8 ------------------
__global__ __launch_bounds__(256) void edgeconv1_kernel(
    const _Float16* __restrict__ svF, const _Float16* __restrict__ trkF,
    const _Float16* __restrict__ WaT, const _Float16* __restrict__ WbT,
    const float* __restrict__ bc, _Float16* __restrict__ f1F) {
  __shared__ _Float16 trkL[64 * 136];
  __shared__ _Float16 svL[16 * 136];
  __shared__ _Float16 P[64 * 136];    // trk @ (Wtop-Wbot)
  __shared__ _Float16 Q[16 * 136];    // sv  @  Wbot
  __shared__ float d2[64 * 16];
  __shared__ float ntrk[64];
  __shared__ float nsv[16];
  __shared__ short knn[64 * 8];
  int tid = threadIdx.x, lane = tid & 31, wave = tid >> 5;
  long g = blockIdx.x;
  const _Float16* trkG = trkF + g * 64 * 128;
  const _Float16* svG = svF + g * 16 * 128;

  for (int i = tid; i < 64 * 16; i += 256) {
    int r = i >> 4, c8 = (i & 15) << 3;
    *(v8h*)&trkL[r * 136 + c8] = *(const v8h*)&trkG[r * 128 + c8];
  }
  for (int i = tid; i < 16 * 16; i += 256) {
    int r = i >> 4, c8 = (i & 15) << 3;
    *(v8h*)&svL[r * 136 + c8] = *(const v8h*)&svG[r * 128 + c8];
  }
  __syncthreads();

  if (wave < 4) {  // dots = trk @ sv^T : 4 m-tiles x 1 n-tile (wave-uniform)
    v8f c = {};
#pragma unroll
    for (int kc = 0; kc < 4; ++kc) {
      v16h a = load_fragA(&trkL[(wave * 16) * 136 + kc * 32], 136, lane);
      v16h b = load_fragB(&svL[kc * 32], 136, lane);
      c = wmma16(a, b, c);
    }
    int s = lane & 15, mh = (lane >> 4) << 3;
#pragma unroll
    for (int j = 0; j < 8; ++j) d2[(wave * 16 + j + mh) * 16 + s] = c[j];
  } else {  // squared norms
    int w = wave - 4;
    if (w < 2) {
      int q = w * 32 + lane;
      ntrk[q] = row_sumsq(&trkL[q * 136], 16);
    } else if (w == 2 && lane < 16) {
      nsv[lane] = row_sumsq(&svL[lane * 136], 16);
    }
  }
  __syncthreads();
  for (int i = tid; i < 64 * 16; i += 256) {
    int q = i >> 4, s = i & 15;
    d2[i] = ntrk[q] - 2.0f * d2[i] + nsv[s];
  }
  __syncthreads();
  if (tid < 64) {  // top-8 (first-index tie-break, matches lax.top_k)
    int q = tid;
    for (int k = 0; k < 8; ++k) {
      float best = 3.0e38f; int bi = 0;
      for (int s = 0; s < 16; ++s) {
        float v = d2[q * 16 + s];
        if (v < best) { best = v; bi = s; }
      }
      knn[q * 8 + k] = (short)bi;
      d2[q * 16 + bi] = 3.0e38f;
    }
  }
  __syncthreads();
  // P (32 tiles) and Q (8 tiles), wave-uniform tile assignment
  for (int t = wave; t < 40; t += 8) {
    const _Float16* Ab; _Float16* Dst; const _Float16* Wt; int mrow, nt;
    if (t < 32) { int mt = t >> 3; Ab = &trkL[(mt * 16) * 136]; Dst = P; Wt = WaT; mrow = mt * 16; nt = t & 7; }
    else        { Ab = &svL[0]; Dst = Q; Wt = WbT; mrow = 0; nt = t - 32; }
    v8f c = {};
#pragma unroll
    for (int kc = 0; kc < 4; ++kc) {
      v16h a = load_fragA(Ab + kc * 32, 136, lane);
      v16h b = load_fragB(Wt + nt * 16 * 128 + kc * 32, 128, lane);
      c = wmma16(a, b, c);
    }
    int n = nt * 16 + (lane & 15), mh = (lane >> 4) << 3;
#pragma unroll
    for (int j = 0; j < 8; ++j) Dst[(mrow + j + mh) * 136 + n] = (_Float16)c[j];
  }
  __syncthreads();
  // msg = elu(P[q] + Q[knn] + b); f1 = max_k msg
  for (int i = tid; i < 64 * 128; i += 256) {
    int q = i >> 7, n = i & 127;
    float p = (float)P[q * 136 + n] + bc[n];
    float best = -3.0e38f;
#pragma unroll
    for (int k = 0; k < 8; ++k) {
      int s = knn[q * 8 + k];
      best = fmaxf(best, elu1(p + (float)Q[s * 136 + n]));
    }
    f1F[(g * 64 + q) * 128 + n] = (_Float16)best;
  }
}

// ---------------- EdgeConv 2 + mean pool + output head ----------------------
__global__ __launch_bounds__(256) void edgeconv2_kernel(
    const _Float16* __restrict__ trkF, const _Float16* __restrict__ f1F,
    const _Float16* __restrict__ WaT, const _Float16* __restrict__ WbT,
    const float* __restrict__ bc, const float* __restrict__ Wo1,
    const float* __restrict__ bo1, const float* __restrict__ Wo2,
    const float* __restrict__ bo2, const float* __restrict__ Wo3,
    const float* __restrict__ bo3, const float* __restrict__ Wo4,
    const float* __restrict__ bo4, float* __restrict__ out) {
  __shared__ _Float16 trkL[64 * 136];
  __shared__ _Float16 srcL[64 * 136];
  __shared__ _Float16 P[64 * 136];
  __shared__ float d2Q[64 * 68];  // d2 (64x64 f32), later reused as Q (64x136 f16)
  __shared__ float ntrk[64], nsrc[64];
  __shared__ short knn[64 * 8];
  __shared__ float partial[256];
  __shared__ float pooled[128];
  __shared__ float h1[64], h2[32], h3[4];
  int tid = threadIdx.x, lane = tid & 31, wave = tid >> 5;
  long g = blockIdx.x;
  const _Float16* trkG = trkF + g * 64 * 128;
  const _Float16* srcG = f1F + g * 64 * 128;

  for (int i = tid; i < 64 * 16; i += 256) {
    int r = i >> 4, c8 = (i & 15) << 3;
    *(v8h*)&trkL[r * 136 + c8] = *(const v8h*)&trkG[r * 128 + c8];
    *(v8h*)&srcL[r * 136 + c8] = *(const v8h*)&srcG[r * 128 + c8];
  }
  __syncthreads();
  if (tid < 64) {
    ntrk[tid] = row_sumsq(&trkL[tid * 136], 16);
  } else if (tid < 128) {
    int q = tid - 64;
    nsrc[q] = row_sumsq(&srcL[q * 136], 16);
  }
  __syncthreads();
  // dots + d2: 4x4 tiles over 8 waves
  for (int t = wave; t < 16; t += 8) {
    int mt = t >> 2, nt = t & 3;
    v8f c = {};
#pragma unroll
    for (int kc = 0; kc < 4; ++kc) {
      v16h a = load_fragA(&trkL[(mt * 16) * 136 + kc * 32], 136, lane);
      v16h b = load_fragB(&srcL[(nt * 16) * 136 + kc * 32], 136, lane);
      c = wmma16(a, b, c);
    }
    int s = nt * 16 + (lane & 15), mh = (lane >> 4) << 3;
#pragma unroll
    for (int j = 0; j < 8; ++j) {
      int q = mt * 16 + j + mh;
      d2Q[q * 64 + s] = ntrk[q] - 2.0f * c[j] + nsrc[s];
    }
  }
  __syncthreads();
  if (tid < 64) {
    int q = tid;
    for (int k = 0; k < 8; ++k) {
      float best = 3.0e38f; int bi = 0;
      for (int s = 0; s < 64; ++s) {
        float v = d2Q[q * 64 + s];
        if (v < best) { best = v; bi = s; }
      }
      knn[q * 8 + k] = (short)bi;
      d2Q[q * 64 + bi] = 3.0e38f;
    }
  }
  __syncthreads();
  _Float16* Q = (_Float16*)d2Q;  // overlay: d2 is dead past here
  for (int t = wave; t < 64; t += 8) {
    int half = t >> 5, tt = t & 31, mt = tt >> 3, nt = tt & 7;
    const _Float16* Ab = half ? &srcL[(mt * 16) * 136] : &trkL[(mt * 16) * 136];
    const _Float16* Wt = half ? WbT : WaT;
    _Float16* Dst = half ? Q : P;
    v8f c = {};
#pragma unroll
    for (int kc = 0; kc < 4; ++kc) {
      v16h a = load_fragA(Ab + kc * 32, 136, lane);
      v16h b = load_fragB(Wt + nt * 16 * 128 + kc * 32, 128, lane);
      c = wmma16(a, b, c);
    }
    int n = nt * 16 + (lane & 15), mh = (lane >> 4) << 3;
#pragma unroll
    for (int j = 0; j < 8; ++j) Dst[(mt * 16 + j + mh) * 136 + n] = (_Float16)c[j];
  }
  __syncthreads();
  // gather + max + deterministic mean pooling (2 threads per channel)
  {
    int n = tid & 127;
    float bias = bc[n];
    float acc = 0.0f;
    for (int i = tid; i < 64 * 128; i += 256) {
      int q = i >> 7;
      float p = (float)P[q * 136 + n] + bias;
      float best = -3.0e38f;
#pragma unroll
      for (int k = 0; k < 8; ++k) {
        int s = knn[q * 8 + k];
        best = fmaxf(best, elu1(p + (float)Q[s * 136 + n]));
      }
      acc += best;
    }
    partial[tid] = acc;
  }
  __syncthreads();
  if (tid < 128) pooled[tid] = (partial[tid] + partial[tid + 128]) * (1.0f / 64.0f);
  __syncthreads();
  if (tid < 64) {
    float s = bo1[tid];
    for (int i = 0; i < 128; ++i) s += pooled[i] * Wo1[i * 64 + tid];
    h1[tid] = elu1(s);
  }
  __syncthreads();
  if (tid < 32) {
    float s = bo2[tid];
    for (int i = 0; i < 64; ++i) s += h1[i] * Wo2[i * 32 + tid];
    h2[tid] = elu1(s);
  }
  __syncthreads();
  if (tid < 4) {
    float s = bo3[tid];
    for (int i = 0; i < 32; ++i) s += h2[i] * Wo3[i * 4 + tid];
    h3[tid] = elu1(s);
  }
  __syncthreads();
  if (tid == 0) {
    float s = bo4[0];
    for (int i = 0; i < 4; ++i) s += h3[i] * Wo4[i];
    out[g] = s;                  // out (G,1)
    out[1024 + g] = (float)g;    // batch = arange(G)
  }
}

// ---------------- launch -----------------------------------------------------
extern "C" void kernel_launch(void* const* d_in, const int* in_sizes, int n_in,
                              void* d_out, int out_size, void* d_ws,
                              size_t ws_size, hipStream_t stream) {
  (void)in_sizes; (void)n_in; (void)out_size; (void)ws_size;
  const float* x_sv   = (const float*)d_in[0];
  const float* x_trk  = (const float*)d_in[1];
  const float* W_sv1  = (const float*)d_in[2];
  const float* b_sv1  = (const float*)d_in[3];
  const float* W_sv2  = (const float*)d_in[4];
  const float* b_sv2  = (const float*)d_in[5];
  const float* W_trk1 = (const float*)d_in[6];
  const float* b_trk1 = (const float*)d_in[7];
  const float* W_trk2 = (const float*)d_in[8];
  const float* b_trk2 = (const float*)d_in[9];
  const float* W_c1   = (const float*)d_in[10];
  const float* b_c1   = (const float*)d_in[11];
  const float* W_c2   = (const float*)d_in[12];
  const float* b_c2   = (const float*)d_in[13];
  const float* W_o1   = (const float*)d_in[14];
  const float* b_o1   = (const float*)d_in[15];
  const float* W_o2   = (const float*)d_in[16];
  const float* b_o2   = (const float*)d_in[17];
  const float* W_o3   = (const float*)d_in[18];
  const float* b_o3   = (const float*)d_in[19];
  const float* W_o4   = (const float*)d_in[20];
  const float* b_o4   = (const float*)d_in[21];

  char* w = (char*)d_ws;
  auto carve = [&](size_t bytes) -> char* {
    char* p = w; w += (bytes + 255) & ~(size_t)255; return p;
  };
  _Float16* svF    = (_Float16*)carve((size_t)16384 * 128 * 2);  // 4 MB
  _Float16* trkF   = (_Float16*)carve((size_t)65536 * 128 * 2);  // 16 MB
  _Float16* f1F    = (_Float16*)carve((size_t)65536 * 128 * 2);  // 16 MB
  _Float16* WsvT1  = (_Float16*)carve(128 * 32 * 2);
  _Float16* WtrkT1 = (_Float16*)carve(128 * 32 * 2);
  _Float16* WsvT2  = (_Float16*)carve(128 * 128 * 2);
  _Float16* WtrkT2 = (_Float16*)carve(128 * 128 * 2);
  _Float16* Wc1aT  = (_Float16*)carve(128 * 128 * 2);
  _Float16* Wc1bT  = (_Float16*)carve(128 * 128 * 2);
  _Float16* Wc2aT  = (_Float16*)carve(128 * 128 * 2);
  _Float16* Wc2bT  = (_Float16*)carve(128 * 128 * 2);

  transpose_pad_f16<<<16, 256, 0, stream>>>(W_sv1, 2, 128, 32, WsvT1);
  transpose_pad_f16<<<16, 256, 0, stream>>>(W_trk1, 8, 128, 32, WtrkT1);
  transpose_pad_f16<<<64, 256, 0, stream>>>(W_sv2, 128, 128, 128, WsvT2);
  transpose_pad_f16<<<64, 256, 0, stream>>>(W_trk2, 128, 128, 128, WtrkT2);
  combine_edge_f16<<<64, 256, 0, stream>>>(W_c1, Wc1aT, Wc1bT);
  combine_edge_f16<<<64, 256, 0, stream>>>(W_c2, Wc2aT, Wc2bT);

  mlp2_kernel<<<128, 256, 0, stream>>>(x_sv, 2, WsvT1, b_sv1, WsvT2, b_sv2, svF);
  mlp2_kernel<<<512, 256, 0, stream>>>(x_trk, 8, WtrkT1, b_trk1, WtrkT2, b_trk2, trkF);

  edgeconv1_kernel<<<1024, 256, 0, stream>>>(svF, trkF, Wc1aT, Wc1bT, b_c1, f1F);
  edgeconv2_kernel<<<1024, 256, 0, stream>>>(trkF, f1F, Wc2aT, Wc2bT, b_c2,
                                             W_o1, b_o1, W_o2, b_o2, W_o3, b_o3,
                                             W_o4, b_o4, (float*)d_out);
}
